// ContrastiveLoss_12163347382790
// MI455X (gfx1250) — matrix operand order repaired and checked by reading, not compiled, not run on hardware
//
#include <hip/hip_runtime.h>

// ---------------------------------------------------------------------------
// PatchNCE-style contrastive loss, fused for MI455X (gfx1250, wave32, WMMA).
// B=4, C=256, N=4096. GEMM in bf16 WMMA (raw data), exact f32 L1 scales after.
// Round 3: async global->LDS staging via inline asm (ASYNCcnt), double-buffered
// k tiles so the DMA of tile t+1 overlaps the 8-WMMA chain on tile t.
// ---------------------------------------------------------------------------

typedef __attribute__((ext_vector_type(8)))  __bf16 v8bf;
typedef __attribute__((ext_vector_type(16))) __bf16 v16bf;
typedef __attribute__((ext_vector_type(8)))  float  v8f;

constexpr int   B_    = 4;
constexpr int   C_    = 256;
constexpr int   N_    = 4096;
constexpr float INV_T = 1.0f / 0.07f;   // also the fixed logsumexp max M0
constexpr float EPS_  = 1e-12f;

// ---------------------------------------------------------------------------
// Kernel 1: per-(b,n) L1 norms of q and k (reads coalesced across n).
// invq carries the 1/T factor so the GEMM scale is invq[row]*invk[col].
// ---------------------------------------------------------------------------
__global__ __launch_bounds__(256) void nce_norms(const float* __restrict__ fq,
                                                 const float* __restrict__ fk,
                                                 float* __restrict__ invq,
                                                 float* __restrict__ invk) {
  int t = blockIdx.x * 256 + threadIdx.x;          // [0, B*N)
  int b = t >> 12;
  int n = t & (N_ - 1);
  const float* qp = fq + (size_t)b * C_ * N_ + n;
  const float* kp = fk + (size_t)b * C_ * N_ + n;
  float sq = 0.f, sk = 0.f;
#pragma unroll 4
  for (int c = 0; c < C_; ++c) {
    sq += __builtin_fabsf(qp[(size_t)c * N_]);
    sk += __builtin_fabsf(kp[(size_t)c * N_]);
  }
  invq[t] = INV_T / fmaxf(sq, EPS_);
  invk[t] = 1.0f  / fmaxf(sk, EPS_);
}

// ---------------------------------------------------------------------------
// Kernel 2: LDS-tiled transpose+convert  [B,C,N] f32  ->  [B,N,C] bf16.
// ---------------------------------------------------------------------------
__global__ __launch_bounds__(256) void nce_tr_bf16(const float* __restrict__ src,
                                                   __bf16* __restrict__ dst) {
  __shared__ float tile[32][33];
  int n0 = blockIdx.x * 32, c0 = blockIdx.y * 32, b = blockIdx.z;
  int tx = threadIdx.x, ty = threadIdx.y;
#pragma unroll
  for (int i = 0; i < 4; ++i) {
    int c = c0 + ty + i * 8;
    tile[ty + i * 8][tx] = src[((size_t)b * C_ + c) * N_ + n0 + tx];
  }
  __syncthreads();
#pragma unroll
  for (int i = 0; i < 4; ++i) {
    int n = n0 + ty + i * 8;
    dst[((size_t)b * N_ + n) * C_ + c0 + tx] = (__bf16)tile[tx][ty + i * 8];
  }
}

// ---------------------------------------------------------------------------
// Async global->LDS staging (gfx1250 GLOBAL_LOAD_ASYNC_TO_LDS_B128, ASYNCcnt)
// via inline asm: VDST = LDS byte-address VGPR (low 32 bits of the generic
// shared pointer), VADDR = 64-bit global address pair, SADDR = off (GV mode).
// Set NCE_ASYNC_LDS to 0 to fall back to a VGPR-bounce copy.
// ---------------------------------------------------------------------------
#define NCE_ASYNC_LDS 1

__device__ __forceinline__ void nce_async_wait0() {
#if __has_builtin(__builtin_amdgcn_s_wait_asynccnt)
  __builtin_amdgcn_s_wait_asynccnt(0);
#else
  asm volatile("s_wait_asynccnt 0x0" ::: "memory");
#endif
}

// Stage one 16-row k tile (8 KB, contiguous) into LDS; 128 threads, 64B each.
__device__ __forceinline__ void nce_stage_tile(const __bf16* __restrict__ gsrc,
                                               __bf16* ldst, int tid) {
#if NCE_ASYNC_LDS
  const char* g = (const char*)gsrc + tid * 16;
  uint32_t    l = (uint32_t)(uintptr_t)ldst + (uint32_t)tid * 16u;
#pragma unroll
  for (int i = 0; i < 4; ++i) {
    uint64_t ga = (uint64_t)(uintptr_t)(g + i * 2048);
    uint32_t la = l + (uint32_t)(i * 2048);
    asm volatile("global_load_async_to_lds_b128 %0, %1, off"
                 :: "v"(la), "v"(ga)
                 : "memory");
  }
#else
  const uint4* gs = (const uint4*)gsrc;
  uint4* ls = (uint4*)ldst;
#pragma unroll
  for (int i = 0; i < 4; ++i) ls[tid + i * 128] = gs[tid + i * 128];
#endif
}

// ---------------------------------------------------------------------------
// Kernel 3: fused GEMM (bf16 WMMA, K=256) + streaming logsumexp.
// Grid: B*(N/64) blocks of 128 threads (4 waves). Wave w owns rows
// n0 = rg*64 + w*16. k tiles are double-buffered in LDS: async stage of
// tile t+1 overlaps the 8-WMMA chain on tile t; one barrier per tile.
// ---------------------------------------------------------------------------
__global__ __launch_bounds__(128) void nce_gemm_lse(const __bf16* __restrict__ q16,
                                                    const __bf16* __restrict__ k16,
                                                    const float* __restrict__ invq,
                                                    const float* __restrict__ invk,
                                                    float* __restrict__ partial) {
  __shared__ __align__(16) __bf16 ktile[2][16 * C_];  // 2 x 8 KB double buffer
  __shared__ float wsum[4];

  const int tid  = threadIdx.x;
  const int w    = tid >> 5;
  const int lane = tid & 31;
  const int r    = lane & 15;
  const int hi   = lane >> 4;

  const int blk = blockIdx.x;
  const int b   = blk >> 6;          // 64 row-groups per batch
  const int rg  = blk & 63;
  const int n0  = rg * 64 + w * 16;  // this wave's 16-row base
  const size_t bN = (size_t)b * N_;

  // --- Preload A fragments for 16 rows x K=256 (contiguous k order, matching B:
  // half-wave 0 holds K {0..15}+kc, half-wave 1 holds K {16..31}+kc).
  v16bf a_frag[8];
  {
    const __bf16* qb = q16 + (bN + n0 + r) * C_ + (hi ? 16 : 0);
#pragma unroll
    for (int kc = 0; kc < 8; ++kc) {
      v8bf lo = *(const v8bf*)(qb + kc * 32);
      v8bf hh = *(const v8bf*)(qb + kc * 32 + 8);
      a_frag[kc] = __builtin_shufflevector(lo, hh, 0, 1, 2, 3, 4, 5, 6, 7,
                                           8, 9, 10, 11, 12, 13, 14, 15);
    }
  }
  float invqv[8];
#pragma unroll
  for (int v = 0; v < 8; ++v) invqv[v] = invq[bN + n0 + (hi << 3) + v];

  float s8[8] = {};       // streaming sum of exp(val - INV_T) per owned row
  float diagval = 0.f;    // l_pos/T for the row this lane owns (if any)

  // Prologue: stage tile 0.
  nce_stage_tile(k16 + bN * C_, &ktile[0][0], tid);
  nce_async_wait0();
  __syncthreads();

  int cur = 0;
  for (int m0 = 0; m0 < N_; m0 += 16) {
    // Kick off async stage of the next tile into the other buffer.
    if (m0 + 16 < N_)
      nce_stage_tile(k16 + (bN + m0 + 16) * C_, &ktile[cur ^ 1][0], tid);

    // Lift all 8 B fragments (16x ds_load_b128) ahead of the WMMA chain.
    const __bf16* kb = &ktile[cur][0] + r * C_ + hi * 16;  // B: seq per half
    v16bf bfr[8];
#pragma unroll
    for (int kc = 0; kc < 8; ++kc) bfr[kc] = *(const v16bf*)(kb + kc * 32);

    v8f acc = {};
#pragma unroll
    for (int kc = 0; kc < 8; ++kc)
      acc = __builtin_amdgcn_wmma_f32_16x16x32_bf16(
          false, a_frag[kc], false, bfr[kc], (short)0, acc, false, false);

    // Scale + branch-free streaming sum-exp (|val| <= 1/T by L1-norm bound).
    const float ck = invk[bN + m0 + r];
#pragma unroll
    for (int v = 0; v < 8; ++v) {
      float val = acc[v] * invqv[v] * ck;        // S[n,m]/T in f32
      float e   = __expf(val - INV_T);
      bool isd  = (m0 + r) == (n0 + (hi << 3) + v);
      if (isd) { diagval = val; e = 0.f; }       // masked diag -> exp == 0
      s8[v] += e;
    }

    nce_async_wait0();   // next tile fully landed in LDS
    __syncthreads();     // everyone done reading cur + staging next
    cur ^= 1;
  }

  // Cross-lane reduce within each 16-lane half (rows 0-7 / 8-15).
#pragma unroll
  for (int v = 0; v < 8; ++v)
    for (int mask = 1; mask < 16; mask <<= 1)
      s8[v] += __shfl_xor(s8[v], mask, 32);

  // Fold in the l_pos column, finish logsumexp, subtract l_pos/T.
  float rsum = 0.f;
#pragma unroll
  for (int v = 0; v < 8; ++v) {
    float d = __shfl(diagval, (hi ? 24 : 0) + v, 32);  // owner lanes
    float S = s8[v] + __expf(d - INV_T);
    rsum += INV_T + __logf(S) - d;
  }
  rsum += __shfl_xor(rsum, 16, 32);   // combine the two half-wave row sets
  if (lane == 0) wsum[w] = rsum;
  __syncthreads();
  if (tid == 0) partial[blk] = wsum[0] + wsum[1] + wsum[2] + wsum[3];
}

// ---------------------------------------------------------------------------
// Kernel 4: deterministic fixed-order reduction of 256 partials -> mean.
// ---------------------------------------------------------------------------
__global__ __launch_bounds__(256) void nce_finalize(const float* __restrict__ partial,
                                                    float* __restrict__ out) {
  __shared__ float buf[256];
  int t = threadIdx.x;
  buf[t] = partial[t];
#pragma unroll
  for (int off = 128; off > 0; off >>= 1) {
    __syncthreads();
    if (t < off) buf[t] += buf[t + off];
  }
  if (t == 0) out[0] = buf[0] * (1.0f / 16384.0f);   // mean over B*N rows
}

// ---------------------------------------------------------------------------
extern "C" void kernel_launch(void* const* d_in, const int* in_sizes, int n_in,
                              void* d_out, int out_size, void* d_ws, size_t ws_size,
                              hipStream_t stream) {
  (void)in_sizes; (void)n_in; (void)out_size; (void)ws_size;
  const float* fq = (const float*)d_in[0];
  const float* fk = (const float*)d_in[1];
  float* out = (float*)d_out;

  char* ws = (char*)d_ws;
  const size_t QB = (size_t)B_ * N_ * C_ * sizeof(__bf16);   // 8 MB each
  __bf16* q16   = (__bf16*)ws;
  __bf16* k16   = (__bf16*)(ws + QB);
  float*  invq  = (float*)(ws + 2 * QB);
  float*  invk  = (float*)(ws + 2 * QB + (size_t)B_ * N_ * sizeof(float));
  float*  part  = (float*)(ws + 2 * QB + 2 * (size_t)B_ * N_ * sizeof(float));

  nce_norms<<<(B_ * N_) / 256, 256, 0, stream>>>(fq, fk, invq, invk);
  nce_tr_bf16<<<dim3(N_ / 32, C_ / 32, B_), dim3(32, 8), 0, stream>>>(fq, q16);
  nce_tr_bf16<<<dim3(N_ / 32, C_ / 32, B_), dim3(32, 8), 0, stream>>>(fk, k16);
  nce_gemm_lse<<<B_ * (N_ / 64), 128, 0, stream>>>(q16, k16, invq, invk, part);
  nce_finalize<<<1, 256, 0, stream>>>(part, out);
}
